// AutoRegressiveLSTM_19636590478174
// MI455X (gfx1250) — matrix operand order repaired and checked by reading, not compiled, not run on hardware
//
#include <hip/hip_runtime.h>

// Problem dims (fixed by reference)
#define T_STEPS 1024
#define BATCH   512
#define IN_DIM  128
#define HID     256
#define KDIM    384     // IN_DIM + HID (z handled as rank-1 update)
#define KTILES  12      // KDIM / 32
#define APITCH  392     // padded LDS row pitch (ushorts) -> 784B, avoids bank conflicts
#define NWG     256     // 16 batch tiles x 16 column tiles
#define BT_TILES 16
#define CT_TILES 16

// CDNA5 async global->LDS path (guarded: falls back to load+store if absent)
#if defined(__has_builtin)
#if __has_builtin(__builtin_amdgcn_global_load_async_to_lds_b128) && \
    __has_builtin(__builtin_amdgcn_s_wait_asynccnt)
#define USE_ASYNC_LDS 1
#endif
#if __has_builtin(__builtin_amdgcn_sched_barrier)
#define SCHED_FENCE() __builtin_amdgcn_sched_barrier(0)
#endif
#endif
#ifndef USE_ASYNC_LDS
#define USE_ASYNC_LDS 0
#endif
#ifndef SCHED_FENCE
#define SCHED_FENCE()
#endif

#define AS1 __attribute__((address_space(1)))
#define AS3 __attribute__((address_space(3)))

typedef __attribute__((ext_vector_type(16))) __bf16 v16bf;
typedef __attribute__((ext_vector_type(8)))  float  v8f;
typedef int v4i_g __attribute__((vector_size(16)));   // matches builtin param type

struct alignas(16) U4 { unsigned x, y, z, w; };
struct U8x { U4 lo, hi; };                  // 32B = one v16bf fragment
struct alignas(16) F4 { float x, y, z, w; };

__device__ __forceinline__ unsigned short f2bf(float f) {
  unsigned u = __float_as_uint(f);
  u += 0x7FFFu + ((u >> 16) & 1u);          // round-to-nearest-even
  return (unsigned short)(u >> 16);
}
__device__ __forceinline__ float sigmoidf(float x) {
  return 1.0f / (1.0f + __expf(-x));
}
__device__ __forceinline__ float tanh_fast(float x) {
  x = fminf(15.0f, fmaxf(-15.0f, x));
  float e = __expf(2.0f * x);
  return (e - 1.0f) / (e + 1.0f);
}

__device__ __forceinline__ void grid_sync(unsigned* bar, unsigned nwg) {
  __syncthreads();
  if (threadIdx.x == 0) {
    __threadfence();
    volatile unsigned* vgen = bar + 1;
    unsigned gen = *vgen;
    if (atomicAdd(bar, 1u) == nwg - 1u) {
      bar[0] = 0;
      __threadfence();
      atomicAdd(bar + 1, 1u);
    } else {
      while (*vgen == gen) { __builtin_amdgcn_s_sleep(1); }
    }
    __threadfence();
  }
  __syncthreads();
}

// ---------------------------------------------------------------------------
// Prep 1: pack [W_ih | W_hh] into bf16 WMMA B-fragment order.
// Entry (tileId, kt, lane) = 16 bf16: column r = g*256 + ct*16 + (lane&15),
// K = kt*32 + (lane>>4)*16 + 0..15.
// ---------------------------------------------------------------------------
__global__ void prep_pack(const float* __restrict__ Wih,
                          const float* __restrict__ Whh,
                          unsigned short* __restrict__ wpack) {
  int idx = blockIdx.x * 256 + threadIdx.x;
  if (idx >= 64 * KTILES * 32) return;
  int lane   = idx & 31;
  int kt     = (idx >> 5) % KTILES;
  int tileId = idx / (KTILES * 32);
  int g = tileId >> 4, ct = tileId & 15;
  int n = lane & 15, hi = lane >> 4;
  int r  = g * HID + ct * 16 + n;           // gate row in [0, 1024)
  int kb = kt * 32 + hi * 16;
  unsigned short* d = wpack + (size_t)idx * 16;
  #pragma unroll
  for (int v = 0; v < 16; ++v) {
    int k = kb + v;
    float w = (k < IN_DIM) ? Wih[(size_t)r * (IN_DIM + 1) + k]
                           : Whh[(size_t)r * HID + (k - IN_DIM)];
    d[v] = f2bf(w);
  }
}

// Prep 2: h0 -> bf16 ring slot 0; reset grid barrier.
__global__ void prep_init(const float* __restrict__ h0,
                          unsigned short* __restrict__ hbuf,
                          unsigned* __restrict__ bar) {
  int idx = blockIdx.x * 256 + threadIdx.x;
  if (idx < BATCH * HID) hbuf[idx] = f2bf(h0[idx]);
  if (idx == 0) { bar[0] = 0u; bar[1] = 0u; }
}

// ---------------------------------------------------------------------------
// Persistent recurrent kernel. 256 WGs x 256 threads (8 waves).
// WG (bt, ct): batch rows [bt*32, +32), hidden units [ct*16, +16), all 4 gates.
// Wave w: mt = w&1 (16-row half), g = w>>1 (gate) -> one 16x16 WMMA tile, K=384.
// ---------------------------------------------------------------------------
__global__ __launch_bounds__(256, 1)
void lstm_persistent(const float* __restrict__ xin,
                     const float* __restrict__ c0,
                     const float* __restrict__ z0,
                     const float* __restrict__ Wih,
                     const float* __restrict__ bih,
                     const float* __restrict__ bhh,
                     const float* __restrict__ Wfc,
                     const float* __restrict__ bfcp,
                     const unsigned short* __restrict__ wpack,
                     unsigned short* __restrict__ hbuf,   // [2][B*H] bf16 ring
                     float* __restrict__ zpart,           // [2][B][16]
                     unsigned* __restrict__ bar,
                     float* __restrict__ outz,            // [T*B]
                     float* __restrict__ hout,            // [B*H]
                     float* __restrict__ cout)            // [B*H]
{
  __shared__ unsigned short Apanel[32 * APITCH];  // 25 KB: [x_t | h] bf16 panel
  __shared__ float gbuf[32][4][16];               // gate exchange (8 KB)
  __shared__ float zred[32][16];
  __shared__ float zl[32];
  __shared__ float biasl[4][16];
  __shared__ float wzl[4][16];
  __shared__ float wfcl[16];

  const int tid   = threadIdx.x;
  const int bt    = blockIdx.x & (BT_TILES - 1);
  const int ct    = blockIdx.x >> 4;
  const int bbase = bt * 32;
  const int jbase = ct * 16;
  const int wave  = tid >> 5;
  const int lane  = tid & 31;
  const int mt    = wave & 1;
  const int g     = wave >> 1;
  const float bfc = bfcp[0];

  // Step-invariant LDS: fused bias, z-column of W_ih, W_fc slice.
  if (tid < 64) {
    int gg = tid >> 4, n = tid & 15;
    int r = gg * HID + jbase + n;
    biasl[gg][n] = bih[r] + bhh[r];
    wzl[gg][n]   = Wih[(size_t)r * (IN_DIM + 1) + IN_DIM];
  }
  if (tid < 16) wfcl[tid] = Wfc[jbase + tid];

  // Persistent cell state: 2 elements per thread, f32 in registers.
  float creg[2];
  #pragma unroll
  for (int s = 0; s < 2; ++s) {
    int e = tid + s * 256;
    int row = e >> 4, col = e & 15;
    creg[s] = c0[(size_t)(bbase + row) * HID + jbase + col];
  }

  // Weight fragments: register-resident for the entire sequence (96 VGPRs).
  v16bf bfrag[KTILES];
  {
    int tileId = g * CT_TILES + ct;
    const U8x* wp = (const U8x*)wpack + (size_t)tileId * KTILES * 32 + lane;
    #pragma unroll
    for (int kt = 0; kt < KTILES; ++kt)
      bfrag[kt] = __builtin_bit_cast(v16bf, wp[kt * 32]);
  }

  for (int t = 0; t < T_STEPS; ++t) {
    const int p = t & 1, q = p ^ 1;

    // Finalize z_{t-1} from contiguous per-batch partials (one cacheline,
    // 4x global_load_b128); emit outputs[t-1] once (ct==0 WGs).
    if (tid < 32) {
      int b = bbase + tid;
      float zn;
      if (t == 0) {
        zn = z0[b];
      } else {
        const F4* zp = (const F4*)(zpart + ((size_t)p * BATCH + b) * CT_TILES);
        F4 z0v = zp[0], z1v = zp[1], z2v = zp[2], z3v = zp[3];
        float s = bfc
                + ((z0v.x + z0v.y) + (z0v.z + z0v.w))
                + ((z1v.x + z1v.y) + (z1v.z + z1v.w))
                + ((z2v.x + z2v.y) + (z2v.z + z2v.w))
                + ((z3v.x + z3v.y) + (z3v.z + z3v.w));
        zn = tanh_fast(s);
        if (ct == 0) outz[(size_t)(t - 1) * BATCH + b] = zn;
      }
      zl[tid] = zn;
    }

    // Stage h_t (bf16, L2-resident ring) into A-panel cols [128,384) via the
    // CDNA5 async global->LDS DMA path (no VGPR round trip).
    {
      const unsigned short* hp = hbuf + (size_t)p * BATCH * HID;
      #pragma unroll
      for (int i = 0; i < 4; ++i) {
        int idx = tid + i * 256;
        int row = idx >> 5, c8 = idx & 31;
        const unsigned short* src = hp + (size_t)(bbase + row) * HID + c8 * 8;
        unsigned short* dst = &Apanel[row * APITCH + IN_DIM + c8 * 8];
#if USE_ASYNC_LDS
        __builtin_amdgcn_global_load_async_to_lds_b128(
            (AS1 v4i_g*)src, (AS3 v4i_g*)dst, 0, 0);
#else
        *(U4*)dst = *(const U4*)src;
#endif
      }
    }
    // Stage x_t (f32 -> bf16) into A-panel cols [0,128).
    {
      const float* xt = xin + (size_t)t * BATCH * IN_DIM + (size_t)bbase * IN_DIM;
      #pragma unroll
      for (int i = 0; i < 4; ++i) {
        int idx = tid + i * 256;
        int row = idx >> 5, c4 = idx & 31;
        F4 v = *(const F4*)(xt + row * IN_DIM + c4 * 4);
        unsigned short* d = &Apanel[row * APITCH + c4 * 4];
        d[0] = f2bf(v.x); d[1] = f2bf(v.y); d[2] = f2bf(v.z); d[3] = f2bf(v.w);
      }
    }
    // Prefetch next timestep's x tile into cache (global_prefetch_b8).
    if (t + 1 < T_STEPS && tid < 128) {
      const float* nx = xin + (size_t)(t + 1) * BATCH * IN_DIM
                        + (size_t)(bbase + (tid >> 2)) * IN_DIM + (tid & 3) * 32;
      __builtin_prefetch(nx, 0, 1);
    }
#if USE_ASYNC_LDS
    __builtin_amdgcn_s_wait_asynccnt(0);
#endif
    __syncthreads();

    // WMMA gate GEMM: one 16x16 f32 tile / wave, K = 384 (12 x bf16 k32).
    // 3-buffer A-fragment rotation, 2 fragments in flight. sched_barrier
    // pins "load(k+2) above wmma(k)" so the allocator cannot coalesce the
    // buffers and the wait inserter emits partial dscnt waits (0x4) instead
    // of a full LDS drain before every WMMA.
    {
      const unsigned short* Ar = &Apanel[(mt * 16 + (lane & 15)) * APITCH];
      const int hio = (lane >> 4) * 16;    // A layout: lanes 16-31 take K+16..K+31
      v8f acc = {0.f, 0.f, 0.f, 0.f, 0.f, 0.f, 0.f, 0.f};
      U8x abuf[3];
      abuf[0].lo = *(const U4*)(Ar + hio);
      abuf[0].hi = *(const U4*)(Ar + 8 + hio);
      abuf[1].lo = *(const U4*)(Ar + 32 + hio);
      abuf[1].hi = *(const U4*)(Ar + 40 + hio);
      SCHED_FENCE();
      #pragma unroll
      for (int kt = 0; kt < KTILES; ++kt) {
        if (kt + 2 < KTILES) {
          abuf[(kt + 2) % 3].lo = *(const U4*)(Ar + (kt + 2) * 32 + hio);
          abuf[(kt + 2) % 3].hi = *(const U4*)(Ar + (kt + 2) * 32 + 8 + hio);
        }
        SCHED_FENCE();
        acc = __builtin_amdgcn_wmma_f32_16x16x32_bf16(
            false, __builtin_bit_cast(v16bf, abuf[kt % 3]),
            false, bfrag[kt],
            (short)0, acc, false, false);
        SCHED_FENCE();
      }
      // Scatter C tile (VGPR a: M=a | M=8+a) to LDS for gate exchange.
      int n = lane & 15, mh = (lane >> 4) * 8;
      #pragma unroll
      for (int a8 = 0; a8 < 8; ++a8)
        gbuf[mt * 16 + mh + a8][g][n] = acc[a8];
    }
    __syncthreads();

    // Elementwise cell update: 2 (row, col) elements per thread.
    #pragma unroll
    for (int s = 0; s < 2; ++s) {
      int e = tid + s * 256;
      int row = e >> 4, col = e & 15;
      float zz = zl[row];
      float iv = gbuf[row][0][col] + biasl[0][col] + zz * wzl[0][col];
      float fv = gbuf[row][1][col] + biasl[1][col] + zz * wzl[1][col];
      float gv = gbuf[row][2][col] + biasl[2][col] + zz * wzl[2][col];
      float ov = gbuf[row][3][col] + biasl[3][col] + zz * wzl[3][col];
      float cn = sigmoidf(fv) * creg[s] + sigmoidf(iv) * tanh_fast(gv);
      creg[s]  = cn;
      float hn = sigmoidf(ov) * tanh_fast(cn);
      size_t gi = (size_t)(bbase + row) * HID + jbase + col;
      hbuf[(size_t)q * BATCH * HID + gi] = f2bf(hn);
      zred[row][col] = hn * wfcl[col];
      if (t == T_STEPS - 1) { hout[gi] = hn; cout[gi] = cn; }
    }
    __syncthreads();

    // Per-WG deterministic reduction of z partials -> slot [q][b][ct].
    if (tid < 32) {
      float s = 0.f;
      #pragma unroll
      for (int j = 0; j < 16; ++j) s += zred[tid][j];
      zpart[((size_t)q * BATCH + (bbase + tid)) * CT_TILES + ct] = s;
    }

    grid_sync(bar, NWG);
  }

  // Epilogue: finalize and emit outputs[T-1] (last zpart parity = T&1 = 0).
  if (ct == 0 && tid < 32) {
    int b = bbase + tid;
    const F4* zp = (const F4*)(zpart + ((size_t)(T_STEPS & 1) * BATCH + b) * CT_TILES);
    F4 z0v = zp[0], z1v = zp[1], z2v = zp[2], z3v = zp[3];
    float s = bfc
            + ((z0v.x + z0v.y) + (z0v.z + z0v.w))
            + ((z1v.x + z1v.y) + (z1v.z + z1v.w))
            + ((z2v.x + z2v.y) + (z2v.z + z2v.w))
            + ((z3v.x + z3v.y) + (z3v.z + z3v.w));
    outz[(size_t)(T_STEPS - 1) * BATCH + b] = tanh_fast(s);
  }
}

extern "C" void kernel_launch(void* const* d_in, const int* in_sizes, int n_in,
                              void* d_out, int out_size, void* d_ws, size_t ws_size,
                              hipStream_t stream) {
  const float* xin = (const float*)d_in[0];
  const float* h0  = (const float*)d_in[1];
  const float* c0  = (const float*)d_in[2];
  const float* z0  = (const float*)d_in[3];
  const float* Wih = (const float*)d_in[4];
  const float* Whh = (const float*)d_in[5];
  const float* bih = (const float*)d_in[6];
  const float* bhh = (const float*)d_in[7];
  const float* Wfc = (const float*)d_in[8];
  const float* bfc = (const float*)d_in[9];

  // Workspace layout (~1.38 MB)
  char* ws = (char*)d_ws;
  unsigned short* wpack = (unsigned short*)ws;                       // 786432 B
  unsigned short* hbuf  = (unsigned short*)(ws + 786432);            // 524288 B
  float*          zpart = (float*)(ws + 786432 + 524288);            //  65536 B
  unsigned*       bar   = (unsigned*)(ws + 786432 + 524288 + 65536); //    128 B

  float* outz = (float*)d_out;
  float* hout = outz + (size_t)T_STEPS * BATCH;
  float* cout = hout + (size_t)BATCH * HID;

  prep_pack<<<96, 256, 0, stream>>>(Wih, Whh, wpack);
  prep_init<<<512, 256, 0, stream>>>(h0, hbuf, bar);
  lstm_persistent<<<NWG, 256, 0, stream>>>(xin, c0, z0, Wih, bih, bhh, Wfc, bfc,
                                           wpack, hbuf, zpart, bar,
                                           outz, hout, cout);
}
